// GATNet_48541720379896
// MI455X (gfx1250) — matrix operand order repaired and checked by reading, not compiled, not run on hardware
//
#include <hip/hip_runtime.h>
#include <hip/hip_bf16.h>
#include <float.h>

// ---------------------------------------------------------------------------
// 2-layer GAT for gfx1250 (wave32). Node transforms: v_wmma_f32_16x16x32_bf16,
// one wave computes a 16 x NC row-strip (A fragment loaded once per k-chunk,
// NC/16 accumulators live in VGPRs). Edge softmax/aggregate: streaming
// float-atomic kernels, float4-vectorized on the hot path.
// ---------------------------------------------------------------------------

typedef __attribute__((ext_vector_type(16))) __bf16 v16bf;
typedef __attribute__((ext_vector_type(8)))  float  v8f;

static constexpr int D     = 128;   // embedding dim (= HEADS*HID = layer-2 in dim)
static constexpr int HID   = 32;
static constexpr int HEADS = 4;
static constexpr int OUTC  = 64;
static constexpr float NEG_SLOPE = 0.2f;

#define DEVFN __device__ __forceinline__

DEVFN unsigned short f32_to_bf16(float f) {
    union { float f; unsigned int u; } cv; cv.f = f;
    unsigned int u = cv.u;
    unsigned int r = u + 0x7FFFu + ((u >> 16) & 1u);   // round-to-nearest-even
    return (unsigned short)(r >> 16);
}

DEVFN void atomicMaxF(float* addr, float val) {
    int* ai = (int*)addr;
    int cur = __float_as_int(*addr);
    while (__int_as_float(cur) < val) {
        int old = atomicCAS(ai, cur, __float_as_int(val));
        if (old == cur) break;
        cur = old;
    }
}

DEVFN void edge_nodes(const int* __restrict__ ei, int E, int e, int& s, int& d) {
    if (e < E) { s = ei[e]; d = ei[E + e]; }
    else       { s = d = e - E; }            // appended self-loops
}

// ------------------------- casts / fills -----------------------------------

// 4 channels per thread: float4 gather -> packed 4xbf16 (uint2) store
__global__ void k_gather_cast4(const float* __restrict__ emb,
                               const int* __restrict__ xidx,
                               uint2* __restrict__ out4, int total4) {
    int i = blockIdx.x * blockDim.x + threadIdx.x;
    if (i >= total4) return;
    int n = i / (D / 4), d4 = i % (D / 4);
    float4 v = ((const float4*)(emb + (size_t)xidx[n] * D))[d4];
    union { unsigned short s[4]; uint2 u; } p;
    p.s[0] = f32_to_bf16(v.x); p.s[1] = f32_to_bf16(v.y);
    p.s[2] = f32_to_bf16(v.z); p.s[3] = f32_to_bf16(v.w);
    out4[i] = p.u;
}

// Wt[n*K + k] = bf16(W[k*Nc + n])  (transpose so B-columns are K-contiguous)
__global__ void k_cast_wT(const float* __restrict__ W,
                          unsigned short* __restrict__ Wt, int K, int Nc) {
    int i = blockIdx.x * blockDim.x + threadIdx.x;
    if (i >= K * Nc) return;
    int n = i / K, k = i % K;
    Wt[i] = f32_to_bf16(W[(size_t)k * Nc + n]);
}

__global__ void k_fill(float* __restrict__ p, float v, int n) {
    int i = blockIdx.x * blockDim.x + threadIdx.x;
    if (i < n) p[i] = v;
}

// ------------------------- WMMA GEMM ---------------------------------------
// C[M,NC] = A[M,K](bf16 row-major) @ Bt[NC,K](bf16, B pre-transposed), f32 acc.
// One wave per 16-row strip; NC/16 accumulators; K stepped by 32 via
// v_wmma_f32_16x16x32_bf16.
template<int K, int NC>
__global__ void k_wmma_gemm_bf16(const unsigned short* __restrict__ A,
                                 const unsigned short* __restrict__ Bt,
                                 float* __restrict__ C, int M) {
    constexpr int CT = NC / 16;
    int wave = (blockIdx.x * blockDim.x + threadIdx.x) >> 5;
    int lane = threadIdx.x & 31;
    int rowTiles = (M + 15) >> 4;
    if (wave >= rowTiles) return;                    // wave-uniform exit
    int m0  = wave << 4;
    int l16 = lane & 15;
    int hs  = lane >> 4;

    int arow = m0 + l16; if (arow > M - 1) arow = M - 1;      // data clamp only
    const unsigned int* arowp = (const unsigned int*)(A + (size_t)arow * K);
    const unsigned int* bbase = (const unsigned int*)(Bt + (size_t)l16 * K);

    v8f acc[CT];
#pragma unroll
    for (int ct = 0; ct < CT; ++ct) acc[ct] = (v8f){};

    for (int kt = 0; kt < K; kt += 32) {
        union { v16bf v; unsigned int u[8]; } a;
        const unsigned int* ap = arowp + (kt >> 1);
#pragma unroll
        for (int r = 0; r < 8; ++r) {
            // A 16x32 bf16 layout: reg r, half hs -> K = 2(r&3) + 8*hs + 16*(r>>2)
            int ka = 2 * (r & 3) + 8 * hs + 16 * (r >> 2);
            a.u[r] = ap[ka >> 1];
        }
#pragma unroll
        for (int ct = 0; ct < CT; ++ct) {
            union { v16bf v; unsigned int u[8]; } b;
            const unsigned int* bp = bbase + ((size_t)ct * 16 * K + kt) / 2;
#pragma unroll
            for (int r = 0; r < 8; ++r) {
                // B 32x16 bf16 layout: reg r, half hs -> K = 2(r&3) + 8*hs + 16*(r>>2)
                int kb = 2 * (r & 3) + 8 * hs + 16 * (r >> 2);
                b.u[r] = bp[kb >> 1];
            }
            acc[ct] = __builtin_amdgcn_wmma_f32_16x16x32_bf16(
                false, a.v, false, b.v, (short)0, acc[ct], false, false);
        }
    }

    if (m0 + 16 <= M) {                              // full tile: unguarded
#pragma unroll
        for (int v = 0; v < 8; ++v) {
            float* rp = C + (size_t)(m0 + v + 8 * hs) * NC + l16;
#pragma unroll
            for (int ct = 0; ct < CT; ++ct) rp[ct * 16] = acc[ct][v];
        }
    } else {                                         // partial last tile
#pragma unroll
        for (int v = 0; v < 8; ++v) {
            int row = m0 + v + 8 * hs;
            if (row < M) {
                float* rp = C + (size_t)row * NC + l16;
#pragma unroll
                for (int ct = 0; ct < CT; ++ct) rp[ct * 16] = acc[ct][v];
            }
        }
    }
}

// ------------------------- attention pieces --------------------------------

// alpha_{s,d}[n,h] = sum_c H[n,h,c] * a_{src,dst}[h,c]
__global__ void k_alpha(const float* __restrict__ Hm,
                        const float* __restrict__ a_src, const float* __restrict__ a_dst,
                        float* __restrict__ as, float* __restrict__ ad,
                        int Nn, int H, int C) {
    int i = blockIdx.x * blockDim.x + threadIdx.x;
    if (i >= Nn * H) return;
    int n = i / H, h = i % H;
    const float* hp = Hm + (size_t)n * H * C + (size_t)h * C;
    const float* sp = a_src + h * C;
    const float* dp = a_dst + h * C;
    float vs = 0.f, vd = 0.f;
    for (int c = 0; c < C; ++c) { float x = hp[c]; vs += x * sp[c]; vd += x * dp[c]; }
    as[i] = vs; ad[i] = vd;
}

// leaky-relu logit per (edge,head); segment max via float atomic-max
__global__ void k_edge_logit_max(const int* __restrict__ ei, int E, int Ep,
                                 const float* __restrict__ as, const float* __restrict__ ad,
                                 float* __restrict__ elr, float* __restrict__ m, int H) {
    int i = blockIdx.x * blockDim.x + threadIdx.x;
    if (i >= Ep * H) return;
    int e = i / H, h = i % H;
    int s, d; edge_nodes(ei, E, e, s, d);
    float v = as[(size_t)s * H + h] + ad[(size_t)d * H + h];
    v = v > 0.f ? v : NEG_SLOPE * v;
    elr[i] = v;
    atomicMaxF(&m[(size_t)d * H + h], v);
}

// exp(e - m[dst]) in-place; segment sum of denominators
__global__ void k_edge_expsum(const int* __restrict__ ei, int E, int Ep,
                              float* __restrict__ elr, const float* __restrict__ m,
                              float* __restrict__ den, int H) {
    int i = blockIdx.x * blockDim.x + threadIdx.x;
    if (i >= Ep * H) return;
    int e = i / H, h = i % H;
    int s, d; edge_nodes(ei, E, e, s, d);
    float ex = __expf(elr[i] - m[(size_t)d * H + h]);
    elr[i] = ex;
    atomicAdd(&den[(size_t)d * H + h], ex);
}

// normalize per-(edge,head) coefficient once: coef = ex / den[dst]
__global__ void k_edge_coef(const int* __restrict__ ei, int E, int Ep,
                            float* __restrict__ coef, const float* __restrict__ den, int H) {
    int i = blockIdx.x * blockDim.x + threadIdx.x;
    if (i >= Ep * H) return;
    int e = i / H, h = i % H;
    int s, d; edge_nodes(ei, E, e, s, d);
    coef[i] = coef[i] / den[(size_t)d * H + h];
}

// agg[dst, c..c+3] += H[src, c..c+3] * coef[e,h]  (thread per (edge, 4 channels))
__global__ void k_edge_agg4(const int* __restrict__ ei, int E, int Ep,
                            const float* __restrict__ coef,
                            const float* __restrict__ Hm, float* __restrict__ agg,
                            int H, int C) {
    long long i = (long long)blockIdx.x * blockDim.x + threadIdx.x;
    int HC4 = (H * C) / 4;
    if (i >= (long long)Ep * HC4) return;
    int e  = (int)(i / HC4);
    int c4 = (int)(i % HC4);
    int c  = c4 * 4;
    int h  = c / C;                                  // C % 4 == 0: no head straddle
    int s, d; edge_nodes(ei, E, e, s, d);
    float a = coef[(size_t)e * H + h];
    float4 hv = ((const float4*)(Hm + (size_t)s * H * C))[c4];
    float* ap = agg + (size_t)d * H * C + c;
    atomicAdd(ap + 0, hv.x * a);
    atomicAdd(ap + 1, hv.y * a);
    atomicAdd(ap + 2, hv.z * a);
    atomicAdd(ap + 3, hv.w * a);
}

// layer-1 epilogue: ELU(agg + b1) -> packed bf16 features for layer 2
__global__ void k_finish1v(const float* __restrict__ agg, const float* __restrict__ b,
                           uint2* __restrict__ out4, int total4, int HC4) {
    int i = blockIdx.x * blockDim.x + threadIdx.x;
    if (i >= total4) return;
    int c4 = i % HC4;
    float4 v = ((const float4*)agg)[i];
    const float* bp = b + c4 * 4;
    float e0 = v.x + bp[0], e1 = v.y + bp[1], e2 = v.z + bp[2], e3 = v.w + bp[3];
    e0 = e0 > 0.f ? e0 : (__expf(e0) - 1.f);
    e1 = e1 > 0.f ? e1 : (__expf(e1) - 1.f);
    e2 = e2 > 0.f ? e2 : (__expf(e2) - 1.f);
    e3 = e3 > 0.f ? e3 : (__expf(e3) - 1.f);
    union { unsigned short s[4]; uint2 u; } p;
    p.s[0] = f32_to_bf16(e0); p.s[1] = f32_to_bf16(e1);
    p.s[2] = f32_to_bf16(e2); p.s[3] = f32_to_bf16(e3);
    out4[i] = p.u;
}

__global__ void k_bias_inplace(float* __restrict__ out, const float* __restrict__ b,
                               int Nn, int C) {
    int i = blockIdx.x * blockDim.x + threadIdx.x;
    if (i < Nn * C) out[i] += b[i % C];
}

// ---------------------------------------------------------------------------

extern "C" void kernel_launch(void* const* d_in, const int* in_sizes, int n_in,
                              void* d_out, int out_size, void* d_ws, size_t ws_size,
                              hipStream_t stream) {
    (void)n_in; (void)out_size; (void)ws_size;
    const int*   xidx = (const int*)  d_in[0];
    const int*   ei   = (const int*)  d_in[1];
    const float* emb  = (const float*)d_in[2];
    const float* W1   = (const float*)d_in[3];
    const float* as1  = (const float*)d_in[4];
    const float* ad1  = (const float*)d_in[5];
    const float* b1   = (const float*)d_in[6];
    const float* W2   = (const float*)d_in[7];
    const float* as2  = (const float*)d_in[8];
    const float* ad2  = (const float*)d_in[9];
    const float* b2   = (const float*)d_in[10];

    const int N  = in_sizes[0];
    const int E  = in_sizes[1] / 2;
    const int Ep = E + N;

    // workspace carve-out (256B aligned); buffers reused across the two layers
    char* ws = (char*)d_ws; size_t off = 0;
    auto alloc = [&](size_t bytes) -> void* {
        void* p = ws + off; off = (off + bytes + 255) & ~(size_t)255; return p;
    };
    unsigned short* xb  = (unsigned short*)alloc((size_t)N * D * 2);  // X bf16 / reused as X1 bf16
    unsigned short* w1t = (unsigned short*)alloc((size_t)D * D * 2);  // W1^T bf16
    unsigned short* w2t = (unsigned short*)alloc((size_t)D * OUTC * 2);
    float* h1  = (float*)alloc((size_t)N * D * 4);                    // layer-1 features
    float* asb = (float*)alloc((size_t)N * HEADS * 4);
    float* adb = (float*)alloc((size_t)N * HEADS * 4);
    float* mb  = (float*)alloc((size_t)N * HEADS * 4);
    float* dnb = (float*)alloc((size_t)N * HEADS * 4);
    float* exb = (float*)alloc((size_t)Ep * HEADS * 4);
    float* agg = (float*)alloc((size_t)N * D * 4);                    // agg1 / reused as h2

    auto nb = [](long long n) { return dim3((unsigned)((n + 255) / 256)); };
    const dim3 tb(256);

    // ---- input casts ----
    k_gather_cast4<<<nb((long long)N * D / 4), tb, 0, stream>>>(emb, xidx, (uint2*)xb, N * D / 4);
    k_cast_wT<<<nb(D * D), tb, 0, stream>>>(W1, w1t, D, D);
    k_cast_wT<<<nb(D * OUTC), tb, 0, stream>>>(W2, w2t, D, OUTC);

    // ---- layer 1: H1 = X @ W1 (WMMA bf16, 16x128 strip per wave) ----
    {
        long long waves = (N + 15) / 16;
        k_wmma_gemm_bf16<D, D><<<nb(waves * 32), tb, 0, stream>>>(xb, w1t, h1, N);
    }
    k_alpha<<<nb((long long)N * HEADS), tb, 0, stream>>>(h1, as1, ad1, asb, adb, N, HEADS, HID);
    k_fill<<<nb((long long)N * HEADS), tb, 0, stream>>>(mb, -FLT_MAX, N * HEADS);
    k_fill<<<nb((long long)N * HEADS), tb, 0, stream>>>(dnb, 0.f, N * HEADS);
    k_fill<<<nb((long long)N * D), tb, 0, stream>>>(agg, 0.f, N * D);
    k_edge_logit_max<<<nb((long long)Ep * HEADS), tb, 0, stream>>>(ei, E, Ep, asb, adb, exb, mb, HEADS);
    k_edge_expsum<<<nb((long long)Ep * HEADS), tb, 0, stream>>>(ei, E, Ep, exb, mb, dnb, HEADS);
    k_edge_coef<<<nb((long long)Ep * HEADS), tb, 0, stream>>>(ei, E, Ep, exb, dnb, HEADS);
    k_edge_agg4<<<nb((long long)Ep * (D / 4)), tb, 0, stream>>>(ei, E, Ep, exb, h1, agg, HEADS, HID);
    k_finish1v<<<nb((long long)N * D / 4), tb, 0, stream>>>(agg, b1, (uint2*)xb, N * D / 4, D / 4);

    // ---- layer 2: H2 = X1 @ W2 (WMMA bf16), single head, C=64 ----
    float* h2 = agg;                                   // reuse agg1 storage
    {
        long long waves = (N + 15) / 16;
        k_wmma_gemm_bf16<D, OUTC><<<nb(waves * 32), tb, 0, stream>>>(xb, w2t, h2, N);
    }
    k_alpha<<<nb((long long)N), tb, 0, stream>>>(h2, as2, ad2, asb, adb, N, 1, OUTC);
    k_fill<<<nb((long long)N), tb, 0, stream>>>(mb, -FLT_MAX, N);
    k_fill<<<nb((long long)N), tb, 0, stream>>>(dnb, 0.f, N);
    k_fill<<<nb((long long)N * OUTC), tb, 0, stream>>>((float*)d_out, 0.f, N * OUTC);
    k_edge_logit_max<<<nb((long long)Ep), tb, 0, stream>>>(ei, E, Ep, asb, adb, exb, mb, 1);
    k_edge_expsum<<<nb((long long)Ep), tb, 0, stream>>>(ei, E, Ep, exb, mb, dnb, 1);
    k_edge_coef<<<nb((long long)Ep), tb, 0, stream>>>(ei, E, Ep, exb, dnb, 1);
    k_edge_agg4<<<nb((long long)Ep * (OUTC / 4)), tb, 0, stream>>>(ei, E, Ep, exb, h2, (float*)d_out, 1, OUTC);
    k_bias_inplace<<<nb((long long)N * OUTC), tb, 0, stream>>>((float*)d_out, b2, N, OUTC);
}
